// XLAttention_81673098101656
// MI455X (gfx1250) — compile-verified, matrix operand order checked by reading, not run on hardware
//
#include <hip/hip_runtime.h>
#include <hip/hip_bf16.h>

typedef __attribute__((ext_vector_type(16))) __bf16 v16bf;
typedef __attribute__((ext_vector_type(8)))  float  v8f;

#define D_MODEL 1024
#define N_HEAD  16
#define D_HEAD  64
#define SEQ_L   2048
#define BATCH   2
#define SEQ_J   4096   /* M + L */

// ---------- helpers ----------

__device__ __forceinline__ unsigned short f2bf(float f) {
  unsigned int x = __float_as_uint(f);
  unsigned int r = x + 0x7fffu + ((x >> 16) & 1u);   // round-to-nearest-even
  return (unsigned short)(r >> 16);
}

__device__ __forceinline__ v8f v8f_zero() {
  v8f z;
#pragma unroll
  for (int e = 0; e < 8; ++e) z[e] = 0.0f;
  return z;
}

// Load one 16-lane-wide bf16 WMMA operand fragment (A or B) from a row-major
// source. Per ISA 7.12.2 (16-bit 16x32 A layout): lane l holds row (l&15);
// elements 0..7  <-> K = k0 + (l>>4)*8 + 0..7
// elements 8..15 <-> K = k0 + 16 + (l>>4)*8 + 0..7
// Caller passes rowptr already selected by (lane&15).
__device__ __forceinline__ v16bf ld_frag_row(const unsigned short* rowptr, int k0, int hh) {
  union { v16bf v; uint4 q[2]; } u;
  u.q[0] = *(const uint4*)(rowptr + k0 + hh * 8);
  u.q[1] = *(const uint4*)(rowptr + k0 + 16 + hh * 8);
  return u.v;
}

// ---------- elementwise f32 -> bf16 ----------

__global__ void cvt_bf16_kernel(const float* __restrict__ in,
                                unsigned short* __restrict__ out, int n) {
  int i = blockIdx.x * blockDim.x + threadIdx.x;
  int stride = gridDim.x * blockDim.x;
  for (; i < n; i += stride) out[i] = f2bf(in[i]);
}

// ---------- WMMA GEMM: out[M,N] = X[M,K] @ W[N,K]^T + bias ----------
// One wave computes a 32x64 output tile (2 M-subtiles x 4 N-subtiles).
// Fragment loads are software-pipelined one k-step ahead in distinct
// registers so the s_wait before each WMMA only drains loads issued a full
// iteration earlier. out_f32 != nullptr -> f32 output, else bf16 output.

__global__ __launch_bounds__(128)
void gemm_bf16_kernel(const unsigned short* __restrict__ X,
                      const unsigned short* __restrict__ W,
                      const float* __restrict__ bias,
                      unsigned short* __restrict__ out_bf,
                      float* __restrict__ out_f32,
                      int M, int N, int K) {
  const int lane = threadIdx.x & 31;
  const int wave = blockIdx.x * (blockDim.x >> 5) + (threadIdx.x >> 5);
  const int ntn  = N >> 6;                 // number of 64-wide N tiles
  const int tm   = wave / ntn;
  const int tn   = wave - tm * ntn;
  if (tm * 32 >= M) return;
  const int hh  = lane >> 4;
  const int l15 = lane & 15;
  const int m0 = tm * 32, n0 = tn * 64;

  v8f acc[2][4];
#pragma unroll
  for (int mi = 0; mi < 2; ++mi)
#pragma unroll
    for (int t = 0; t < 4; ++t) acc[mi][t] = v8f_zero();

  const unsigned short* xrow0 = X + (size_t)(m0 + l15) * K;
  const unsigned short* xrow1 = X + (size_t)(m0 + 16 + l15) * K;
  const unsigned short* wrow[4];
#pragma unroll
  for (int t = 0; t < 4; ++t) wrow[t] = W + (size_t)(n0 + t * 16 + l15) * K;

  // prologue: fragments for k0 = 0
  v16bf a0c = ld_frag_row(xrow0, 0, hh);
  v16bf a1c = ld_frag_row(xrow1, 0, hh);
  v16bf bc0 = ld_frag_row(wrow[0], 0, hh);
  v16bf bc1 = ld_frag_row(wrow[1], 0, hh);
  v16bf bc2 = ld_frag_row(wrow[2], 0, hh);
  v16bf bc3 = ld_frag_row(wrow[3], 0, hh);

  for (int k0 = 0; k0 < K - 32; k0 += 32) {
    const int kn = k0 + 32;
    // prefetch next k-step into distinct registers
    v16bf a0n = ld_frag_row(xrow0, kn, hh);
    v16bf a1n = ld_frag_row(xrow1, kn, hh);
    v16bf bn0 = ld_frag_row(wrow[0], kn, hh);
    v16bf bn1 = ld_frag_row(wrow[1], kn, hh);
    v16bf bn2 = ld_frag_row(wrow[2], kn, hh);
    v16bf bn3 = ld_frag_row(wrow[3], kn, hh);

    acc[0][0] = __builtin_amdgcn_wmma_f32_16x16x32_bf16(false, a0c, false, bc0, (short)0, acc[0][0], false, false);
    acc[0][1] = __builtin_amdgcn_wmma_f32_16x16x32_bf16(false, a0c, false, bc1, (short)0, acc[0][1], false, false);
    acc[0][2] = __builtin_amdgcn_wmma_f32_16x16x32_bf16(false, a0c, false, bc2, (short)0, acc[0][2], false, false);
    acc[0][3] = __builtin_amdgcn_wmma_f32_16x16x32_bf16(false, a0c, false, bc3, (short)0, acc[0][3], false, false);
    acc[1][0] = __builtin_amdgcn_wmma_f32_16x16x32_bf16(false, a1c, false, bc0, (short)0, acc[1][0], false, false);
    acc[1][1] = __builtin_amdgcn_wmma_f32_16x16x32_bf16(false, a1c, false, bc1, (short)0, acc[1][1], false, false);
    acc[1][2] = __builtin_amdgcn_wmma_f32_16x16x32_bf16(false, a1c, false, bc2, (short)0, acc[1][2], false, false);
    acc[1][3] = __builtin_amdgcn_wmma_f32_16x16x32_bf16(false, a1c, false, bc3, (short)0, acc[1][3], false, false);

    a0c = a0n; a1c = a1n;
    bc0 = bn0; bc1 = bn1; bc2 = bn2; bc3 = bn3;
  }

  // epilogue: last k-step
  acc[0][0] = __builtin_amdgcn_wmma_f32_16x16x32_bf16(false, a0c, false, bc0, (short)0, acc[0][0], false, false);
  acc[0][1] = __builtin_amdgcn_wmma_f32_16x16x32_bf16(false, a0c, false, bc1, (short)0, acc[0][1], false, false);
  acc[0][2] = __builtin_amdgcn_wmma_f32_16x16x32_bf16(false, a0c, false, bc2, (short)0, acc[0][2], false, false);
  acc[0][3] = __builtin_amdgcn_wmma_f32_16x16x32_bf16(false, a0c, false, bc3, (short)0, acc[0][3], false, false);
  acc[1][0] = __builtin_amdgcn_wmma_f32_16x16x32_bf16(false, a1c, false, bc0, (short)0, acc[1][0], false, false);
  acc[1][1] = __builtin_amdgcn_wmma_f32_16x16x32_bf16(false, a1c, false, bc1, (short)0, acc[1][1], false, false);
  acc[1][2] = __builtin_amdgcn_wmma_f32_16x16x32_bf16(false, a1c, false, bc2, (short)0, acc[1][2], false, false);
  acc[1][3] = __builtin_amdgcn_wmma_f32_16x16x32_bf16(false, a1c, false, bc3, (short)0, acc[1][3], false, false);

#pragma unroll
  for (int mi = 0; mi < 2; ++mi) {
#pragma unroll
    for (int t = 0; t < 4; ++t) {
      const int col = n0 + t * 16 + l15;
      const float bv = bias[col];
#pragma unroll
      for (int v = 0; v < 8; ++v) {
        const int row = m0 + mi * 16 + hh * 8 + v;  // C/D layout: m = half*8 + vgpr
        const float val = acc[mi][t][v] + bv;
        if (out_f32) out_f32[(size_t)row * N + col] = val;
        else         out_bf [(size_t)row * N + col] = f2bf(val);
      }
    }
  }
}

// ---------- vh[j,b, h*64+d] -> vhT[b,h,d, j] ----------

__global__ void transpose_vh_kernel(const unsigned short* __restrict__ vh,
                                    unsigned short* __restrict__ vhT, int total) {
  int o = blockIdx.x * blockDim.x + threadIdx.x;
  int stride = gridDim.x * blockDim.x;
  for (; o < total; o += stride) {
    int j = o & (SEQ_J - 1);
    int rest = o >> 12;            // / SEQ_J
    int d = rest & 63; rest >>= 6;
    int h = rest & 15;
    int b = rest >> 4;
    vhT[o] = vh[((size_t)j * BATCH + b) * D_MODEL + h * D_HEAD + d];
  }
}

// ---------- flash attention ----------
// One wave owns (16-row Q tile, batch b, head h). Streams J in blocks of 32
// with online softmax through per-wave LDS tiles.

__global__ __launch_bounds__(128)
void attn_kernel(const unsigned short* __restrict__ qh,
                 const unsigned short* __restrict__ kh,
                 const unsigned short* __restrict__ vhT,
                 unsigned short* __restrict__ vecout) {
  __shared__ __align__(16) float          s_score[4][16][32];
  __shared__ __align__(16) unsigned short s_prob [4][16][32];
  __shared__ float s_corr[4][16];
  __shared__ float s_sum [4][16];

  const int ws   = threadIdx.x >> 5;
  const int lane = threadIdx.x & 31;
  const int hh   = lane >> 4;
  const int l15  = lane & 15;

  int wave = blockIdx.x * 4 + ws;            // 0..4095
  const int h  = wave & (N_HEAD - 1); wave >>= 4;
  const int b  = wave & (BATCH - 1);
  const int it = wave >> 1;
  const int i0 = it * 16;

  // Q tile: 16 rows x 64 dims -> two A fragments (K=0..31, 32..63)
  const unsigned short* qrow =
      qh + ((size_t)(i0 + l15) * BATCH + b) * D_MODEL + h * D_HEAD;
  const v16bf aq0 = ld_frag_row(qrow, 0, hh);
  const v16bf aq1 = ld_frag_row(qrow, 32, hh);

  v8f o[4];
#pragma unroll
  for (int t = 0; t < 4; ++t) o[t] = v8f_zero();
  float m_r = -3.0e38f, l_r = 0.0f;          // valid in lanes 0..15 (row = lane)

  for (int j0 = 0; j0 < SEQ_J; j0 += 32) {
    // ---- scores: two 16x16 tiles, K=64 over d ----
#pragma unroll
    for (int jt = 0; jt < 2; ++jt) {
      const unsigned short* krow =
          kh + ((size_t)(j0 + jt * 16 + l15) * BATCH + b) * D_MODEL + h * D_HEAD;
      v16bf bk0 = ld_frag_row(krow, 0, hh);
      v16bf bk1 = ld_frag_row(krow, 32, hh);
      v8f s = v8f_zero();
      s = __builtin_amdgcn_wmma_f32_16x16x32_bf16(false, aq0, false, bk0, (short)0, s, false, false);
      s = __builtin_amdgcn_wmma_f32_16x16x32_bf16(false, aq1, false, bk1, (short)0, s, false, false);
#pragma unroll
      for (int v = 0; v < 8; ++v)
        s_score[ws][hh * 8 + v][jt * 16 + l15] = s[v] * 0.125f;  // 1/sqrt(64)
    }
    __syncthreads();

    // ---- online softmax over this 32-key block (one lane per row) ----
    if (lane < 16) {
      const int r = l15;
      float bm = -3.0e38f;
#pragma unroll
      for (int c = 0; c < 32; ++c) bm = fmaxf(bm, s_score[ws][r][c]);
      const float nm = fmaxf(m_r, bm);
      const float corr = __expf(m_r - nm);
      float bs = 0.0f;
#pragma unroll
      for (int c = 0; c < 32; ++c) {
        const float p = __expf(s_score[ws][r][c] - nm);
        s_prob[ws][r][c] = f2bf(p);
        bs += p;
      }
      l_r = l_r * corr + bs;
      m_r = nm;
      s_corr[ws][r] = corr;
    }
    __syncthreads();

    // ---- rescale O accumulators by per-row correction ----
#pragma unroll
    for (int v = 0; v < 8; ++v) {
      const float f = s_corr[ws][hh * 8 + v];
#pragma unroll
      for (int dt = 0; dt < 4; ++dt) o[dt][v] *= f;
    }

    // ---- PV: prob (16x32, A-frag from LDS) x vhT (32x16 per d-subtile) ----
    v16bf pa = ld_frag_row(&s_prob[ws][l15][0], 0, hh);
#pragma unroll
    for (int dt = 0; dt < 4; ++dt) {
      const unsigned short* vrow =
          vhT + (size_t)((b * N_HEAD + h) * D_HEAD + dt * 16 + l15) * SEQ_J;
      v16bf bv = ld_frag_row(vrow, j0, hh);
      o[dt] = __builtin_amdgcn_wmma_f32_16x16x32_bf16(
          false, pa, false, bv, (short)0, o[dt], false, false);
    }
  }

  if (lane < 16) s_sum[ws][l15] = l_r;
  __syncthreads();

#pragma unroll
  for (int v = 0; v < 8; ++v) {
    const int m = hh * 8 + v;
    const float inv = 1.0f / s_sum[ws][m];
    const size_t orow = ((size_t)(i0 + m) * BATCH + b) * D_MODEL + h * D_HEAD;
#pragma unroll
    for (int dt = 0; dt < 4; ++dt)
      vecout[orow + dt * 16 + l15] = f2bf(o[dt][v] * inv);
  }
}

// ---------- launcher ----------

extern "C" void kernel_launch(void* const* d_in, const int* in_sizes, int n_in,
                              void* d_out, int out_size, void* d_ws, size_t ws_size,
                              hipStream_t stream) {
  const float* q   = (const float*)d_in[0];
  const float* kv  = (const float*)d_in[1];
  const float* mem = (const float*)d_in[2];
  const float* w_q = (const float*)d_in[3];
  const float* b_q = (const float*)d_in[4];
  const float* w_k = (const float*)d_in[5];
  const float* b_k = (const float*)d_in[6];
  const float* w_v = (const float*)d_in[7];
  const float* b_v = (const float*)d_in[8];
  const float* w_c = (const float*)d_in[9];
  const float* b_c = (const float*)d_in[10];
  float* out = (float*)d_out;

  unsigned char* ws = (unsigned char*)d_ws;
  size_t off = 0;
  auto take = [&](size_t bytes) {
    unsigned short* p = (unsigned short*)(ws + off);
    off += bytes;
    return p;
  };
  const size_t NQ = (size_t)SEQ_L * BATCH * D_MODEL;   // 4,194,304
  const size_t NC = (size_t)SEQ_J * BATCH * D_MODEL;   // 8,388,608
  const size_t NW = (size_t)D_MODEL * D_MODEL;         // 1,048,576

  unsigned short* qc   = take(NQ * 2);
  unsigned short* cc   = take(NC * 2);
  unsigned short* wqb  = take(NW * 2);
  unsigned short* wkb  = take(NW * 2);
  unsigned short* wvb  = take(NW * 2);
  unsigned short* wcb  = take(NW * 2);
  unsigned short* qhb  = take(NQ * 2);
  unsigned short* khb  = take(NC * 2);
  unsigned short* vhb  = take(NC * 2);
  unsigned short* vhT  = take(NC * 2);
  unsigned short* vecb = take(NQ * 2);
  (void)ws_size; (void)in_sizes; (void)n_in; (void)out_size;

  // f32 -> bf16 conversions (cc = concat(mem, kv) along axis 0)
  cvt_bf16_kernel<<<4096, 256, 0, stream>>>(q,   qc,       (int)NQ);
  cvt_bf16_kernel<<<4096, 256, 0, stream>>>(mem, cc,       (int)NQ);
  cvt_bf16_kernel<<<4096, 256, 0, stream>>>(kv,  cc + NQ,  (int)NQ);
  cvt_bf16_kernel<<<2048, 256, 0, stream>>>(w_q, wqb, (int)NW);
  cvt_bf16_kernel<<<2048, 256, 0, stream>>>(w_k, wkb, (int)NW);
  cvt_bf16_kernel<<<2048, 256, 0, stream>>>(w_v, wvb, (int)NW);
  cvt_bf16_kernel<<<2048, 256, 0, stream>>>(w_c, wcb, (int)NW);

  const int Mq = SEQ_L * BATCH;  // 4096
  const int Mc = SEQ_J * BATCH;  // 8192
  const int blocksQ = (Mq / 32) * (D_MODEL / 64) / 4;  // 512
  const int blocksC = (Mc / 32) * (D_MODEL / 64) / 4;  // 1024

  gemm_bf16_kernel<<<blocksQ, 128, 0, stream>>>(qc, wqb, b_q, qhb, nullptr, Mq, D_MODEL, D_MODEL);
  gemm_bf16_kernel<<<blocksC, 128, 0, stream>>>(cc, wkb, b_k, khb, nullptr, Mc, D_MODEL, D_MODEL);
  gemm_bf16_kernel<<<blocksC, 128, 0, stream>>>(cc, wvb, b_v, vhb, nullptr, Mc, D_MODEL, D_MODEL);

  transpose_vh_kernel<<<8192, 256, 0, stream>>>(vhb, vhT, (int)NC);

  attn_kernel<<<(SEQ_L / 16) * BATCH * N_HEAD / 4, 128, 0, stream>>>(qhb, khb, vhT, vecb);

  gemm_bf16_kernel<<<blocksQ, 128, 0, stream>>>(vecb, wcb, b_c, nullptr, out, Mq, D_MODEL, D_MODEL);
}
